// GIN_PYG_84851373900196
// MI455X (gfx1250) — matrix operand order repaired and checked by reading, not compiled, not run on hardware
//
#include <hip/hip_runtime.h>
#include <hip/hip_bf16.h>

#define N_NODES   100000
#define N_EDGES   1600000
#define FEATS     128
#define N_CLASSES 10
#define NUM_GRAPHS 100
#define BN_EPS    1e-5f

typedef __attribute__((ext_vector_type(16))) __bf16 v16bf;
typedef __attribute__((ext_vector_type(8)))  __bf16 v8bf;
typedef __attribute__((ext_vector_type(4)))  __bf16 v4bf;
typedef __attribute__((ext_vector_type(8)))  float  v8f;

// ---------------- bf16 split helpers (round-to-nearest-even) ----------------
__device__ __forceinline__ unsigned short f32_to_bf16_rne(float x) {
    unsigned int u = __builtin_bit_cast(unsigned int, x);
    unsigned int lsb = (u >> 16) & 1u;
    u += 0x7fffu + lsb;
    return (unsigned short)(u >> 16);
}
__device__ __forceinline__ float bf16_bits_to_f32(unsigned short b) {
    unsigned int u = ((unsigned int)b) << 16;
    return __builtin_bit_cast(float, u);
}
__device__ __forceinline__ void split_bf16(float x, __bf16& hi, __bf16& lo) {
    unsigned short hb = f32_to_bf16_rne(x);
    float hf = bf16_bits_to_f32(hb);
    unsigned short lb = f32_to_bf16_rne(x - hf);
    hi = __builtin_bit_cast(__bf16, hb);
    lo = __builtin_bit_cast(__bf16, lb);
}

// ---------------- init / zero ----------------
__global__ __launch_bounds__(256) void init_copy2_kernel(const float* __restrict__ src,
                                                         float* __restrict__ a,
                                                         float* __restrict__ b, int n4) {
    int i = blockIdx.x * blockDim.x + threadIdx.x;
    if (i >= n4) return;
    float4 v = ((const float4*)src)[i];
    ((float4*)a)[i] = v;
    ((float4*)b)[i] = v;
}

__global__ __launch_bounds__(256) void zero_kernel(float* __restrict__ p, int n) {
    int i = blockIdx.x * blockDim.x + threadIdx.x;
    if (i < n) p[i] = 0.0f;
}

// ---------------- GIN aggregation: Z[dst] += H[src] (Z pre-initialized to H) ----------------
__global__ __launch_bounds__(256) void edge_scatter_kernel(const float* __restrict__ H,
                                                           const int* __restrict__ ei,
                                                           float* __restrict__ Z, int nEdges) {
    int tid  = blockIdx.x * blockDim.x + threadIdx.x;
    int e    = tid >> 5;          // one wave (32 lanes) per edge
    int c    = (tid & 31) * 4;    // 4 consecutive feats per lane
    if (e >= nEdges) return;
    int s = ei[e];
    int d = ei[nEdges + e];
    float4 v = *(const float4*)(H + (size_t)s * FEATS + c);
    float* zp = Z + (size_t)d * FEATS + c;
    atomicAdd(zp + 0, v.x);
    atomicAdd(zp + 1, v.y);
    atomicAdd(zp + 2, v.z);
    atomicAdd(zp + 3, v.w);
}

// ---------------- precompute split operands ----------------
// Wt[n][k] (transposed) hi/lo from W[k][n]: one lane's 16 B-elements become contiguous.
__global__ __launch_bounds__(256) void prep_w_kernel(const float* __restrict__ W,
                                                     __bf16* __restrict__ Wh,
                                                     __bf16* __restrict__ Wl) {
    int i = blockIdx.x * blockDim.x + threadIdx.x;   // 128*128
    if (i >= FEATS * FEATS) return;
    int n = i >> 7, k = i & (FEATS - 1);
    __bf16 h, l;
    split_bf16(W[(size_t)k * FEATS + n], h, l);
    Wh[i] = h;   // Wt[n][k]
    Wl[i] = l;
}

// Elementwise split of Z (row-major) into Ah/Al (same layout, bf16).
__global__ __launch_bounds__(256) void split_a_kernel(const float* __restrict__ Z,
                                                      __bf16* __restrict__ Ah,
                                                      __bf16* __restrict__ Al, int n4) {
    int i = blockIdx.x * blockDim.x + threadIdx.x;
    if (i >= n4) return;
    float4 v = ((const float4*)Z)[i];
    __bf16 h0, l0, h1, l1, h2, l2, h3, l3;
    split_bf16(v.x, h0, l0);
    split_bf16(v.y, h1, l1);
    split_bf16(v.z, h2, l2);
    split_bf16(v.w, h3, l3);
    v4bf h, l;
    h[0] = h0; h[1] = h1; h[2] = h2; h[3] = h3;
    l[0] = l0; l[1] = l1; l[2] = l2; l[3] = l3;
    *(v4bf*)(Ah + (size_t)i * 4) = h;
    *(v4bf*)(Al + (size_t)i * 4) = l;
}

// ---------------- WMMA GEMM: Y = Z @ W + bias  (100000x128 @ 128x128) ----------------
// One wave -> one 16x16 output tile; 8 waves/block cover the 8 column tiles of a
// 16-row stripe. K=128 in 4 steps of 32. fp32 emulated as pre-split bf16 hi/lo:
// 3x v_wmma_f32_16x16x32_bf16 per k-step (hi*hi + hi*lo + lo*hi), f32 accumulate.
__global__ __launch_bounds__(256) void gemm_bias_kernel(const __bf16* __restrict__ Ah,
                                                        const __bf16* __restrict__ Al,
                                                        const __bf16* __restrict__ Wh,
                                                        const __bf16* __restrict__ Wl,
                                                        const float* __restrict__ bias,
                                                        float* __restrict__ Y) {
    const int lane    = threadIdx.x & 31;
    const int wave    = threadIdx.x >> 5;
    const int rowBase = blockIdx.x * 16;
    const int colBase = wave * 16;
    const int halfSel = lane >> 4;     // 0: lanes 0-15, 1: lanes 16-31
    const int lan15   = lane & 15;

    const int kbaseA = halfSel * 16;   // A layout: e -> K = kb+e (contiguous, matches B)
    const int koffB  = halfSel * 16;   // B layout: e -> K = kb+e, col = lane&15
    const int bcol   = colBase + lan15;

    const __bf16* arowH = Ah + (size_t)(rowBase + lan15) * FEATS;
    const __bf16* arowL = Al + (size_t)(rowBase + lan15) * FEATS;
    const __bf16* bH    = Wh + (size_t)bcol * FEATS;
    const __bf16* bL    = Wl + (size_t)bcol * FEATS;

    v8f c = {};
    #pragma unroll
    for (int kk = 0; kk < FEATS; kk += 32) {
        const int ka = kk + kbaseA;
        v8bf ah0 = *(const v8bf*)(arowH + ka);
        v8bf ah1 = *(const v8bf*)(arowH + ka + 8);
        v8bf al0 = *(const v8bf*)(arowL + ka);
        v8bf al1 = *(const v8bf*)(arowL + ka + 8);
        v16bf a_hi = __builtin_shufflevector(ah0, ah1, 0,1,2,3,4,5,6,7,8,9,10,11,12,13,14,15);
        v16bf a_lo = __builtin_shufflevector(al0, al1, 0,1,2,3,4,5,6,7,8,9,10,11,12,13,14,15);
        v16bf b_hi = *(const v16bf*)(bH + kk + koffB);
        v16bf b_lo = *(const v16bf*)(bL + kk + koffB);
        c = __builtin_amdgcn_wmma_f32_16x16x32_bf16(false, a_hi, false, b_hi, (short)0, c, false, false);
        c = __builtin_amdgcn_wmma_f32_16x16x32_bf16(false, a_hi, false, b_lo, (short)0, c, false, false);
        c = __builtin_amdgcn_wmma_f32_16x16x32_bf16(false, a_lo, false, b_hi, (short)0, c, false, false);
    }
    const float bv = bias[bcol];
    #pragma unroll
    for (int r = 0; r < 8; ++r) {   // D layout: VGPR r -> M = r + 8*halfSel, N = lane&15
        int row = rowBase + r + halfSel * 8;
        Y[(size_t)row * FEATS + bcol] = c[r] + bv;
    }
}

// ---------------- BN statistics: per-feature sum / sumsq ----------------
__global__ __launch_bounds__(128) void stats_kernel(const float* __restrict__ Y,
                                                    float* __restrict__ sum,
                                                    float* __restrict__ sumsq,
                                                    int nRows, int rowsPerBlk) {
    int f  = threadIdx.x;
    int r0 = blockIdx.x * rowsPerBlk;
    int r1 = min(r0 + rowsPerBlk, nRows);
    float s = 0.0f, q = 0.0f;
    for (int r = r0; r < r1; ++r) {
        float v = Y[(size_t)r * FEATS + f];
        s += v;
        q += v * v;
    }
    atomicAdd(&sum[f], s);
    atomicAdd(&sumsq[f], q);
}

__global__ __launch_bounds__(128) void bn_finalize_kernel(const float* __restrict__ sum,
                                                          const float* __restrict__ sumsq,
                                                          const float* __restrict__ gamma,
                                                          const float* __restrict__ beta,
                                                          float* __restrict__ scale,
                                                          float* __restrict__ shift, float invN) {
    int f = threadIdx.x;
    float mu   = sum[f] * invN;
    float var  = sumsq[f] * invN - mu * mu;
    float rstd = rsqrtf(var + BN_EPS);
    float sc   = rstd * gamma[f];
    scale[f] = sc;
    shift[f] = beta[f] - mu * sc;
}

// BN + ReLU, write result into H (features) and Z (pre-init for next layer's aggregation)
__global__ __launch_bounds__(256) void bn_relu_kernel(const float* __restrict__ Y,
                                                      const float* __restrict__ scale,
                                                      const float* __restrict__ shift,
                                                      float* __restrict__ H,
                                                      float* __restrict__ Z, int n4) {
    int i = blockIdx.x * blockDim.x + threadIdx.x;
    if (i >= n4) return;
    int col = (i * 4) & (FEATS - 1);
    float4 y  = ((const float4*)Y)[i];
    float4 sc = *(const float4*)(scale + col);
    float4 sh = *(const float4*)(shift + col);
    float4 o;
    o.x = fmaxf(0.0f, y.x * sc.x + sh.x);
    o.y = fmaxf(0.0f, y.y * sc.y + sh.y);
    o.z = fmaxf(0.0f, y.z * sc.z + sh.z);
    o.w = fmaxf(0.0f, y.w * sc.w + sh.w);
    ((float4*)H)[i] = o;
    ((float4*)Z)[i] = o;
}

// ---------------- global mean pool ----------------
__global__ __launch_bounds__(256) void pool_scatter_kernel(const float* __restrict__ H,
                                                           const int* __restrict__ batch,
                                                           float* __restrict__ psum,
                                                           float* __restrict__ pcnt, int nNodes) {
    int tid  = blockIdx.x * blockDim.x + threadIdx.x;
    int node = tid >> 5;
    int c    = (tid & 31) * 4;
    if (node >= nNodes) return;
    int g = batch[node];
    float4 v = *(const float4*)(H + (size_t)node * FEATS + c);
    float* pp = psum + (size_t)g * FEATS + c;
    atomicAdd(pp + 0, v.x);
    atomicAdd(pp + 1, v.y);
    atomicAdd(pp + 2, v.z);
    atomicAdd(pp + 3, v.w);
    if ((tid & 31) == 0) atomicAdd(&pcnt[g], 1.0f);
}

__global__ __launch_bounds__(256) void pool_div_kernel(const float* __restrict__ psum,
                                                       const float* __restrict__ pcnt,
                                                       float* __restrict__ hg) {
    int i = blockIdx.x * blockDim.x + threadIdx.x;
    if (i >= NUM_GRAPHS * FEATS) return;
    int g = i >> 7;
    hg[i] = psum[i] / fmaxf(pcnt[g], 1.0f);
}

// ---------------- MLP head + log_softmax(axis=0) ----------------
__global__ __launch_bounds__(128) void head_kernel(const float* __restrict__ hg,
                                                   const float* __restrict__ fc1w,
                                                   const float* __restrict__ fc1b,
                                                   const float* __restrict__ fc2w,
                                                   const float* __restrict__ fc2b,
                                                   float* __restrict__ out) {
    __shared__ float hg2[NUM_GRAPHS * FEATS];
    __shared__ float logits[NUM_GRAPHS * N_CLASSES];
    int t = threadIdx.x;
    // hg2 = elu(hg @ fc1 + b1); thread t owns output feature t
    for (int g = 0; g < NUM_GRAPHS; ++g) {
        float acc = fc1b[t];
        for (int k = 0; k < FEATS; ++k) acc += hg[g * FEATS + k] * fc1w[k * FEATS + t];
        hg2[g * FEATS + t] = (acc > 0.0f) ? acc : (expf(acc) - 1.0f);
    }
    __syncthreads();
    // logits = hg2 @ fc2 + b2
    for (int i = t; i < NUM_GRAPHS * N_CLASSES; i += 128) {
        int g = i / N_CLASSES, c = i % N_CLASSES;
        float acc = fc2b[c];
        for (int k = 0; k < FEATS; ++k) acc += hg2[g * FEATS + k] * fc2w[k * N_CLASSES + c];
        logits[i] = acc;
    }
    __syncthreads();
    // log_softmax over axis=0 (over graphs), per class
    if (t < N_CLASSES) {
        float m = -3.402823e38f;
        for (int g = 0; g < NUM_GRAPHS; ++g) m = fmaxf(m, logits[g * N_CLASSES + t]);
        float s = 0.0f;
        for (int g = 0; g < NUM_GRAPHS; ++g) s += expf(logits[g * N_CLASSES + t] - m);
        float lse = m + logf(s);
        for (int g = 0; g < NUM_GRAPHS; ++g)
            out[g * N_CLASSES + t] = logits[g * N_CLASSES + t] - lse;
    }
}

// ---------------- host launch ----------------
extern "C" void kernel_launch(void* const* d_in, const int* in_sizes, int n_in,
                              void* d_out, int out_size, void* d_ws, size_t ws_size,
                              hipStream_t stream) {
    const float* h_in   = (const float*)d_in[0];
    const int*   ei     = (const int*)  d_in[2];   // edge_index (2 x N_EDGES)
    const int*   batch  = (const int*)  d_in[3];
    const float* W0     = (const float*)d_in[4];
    const float* Ws     = (const float*)d_in[5];
    const float* bs     = (const float*)d_in[6];
    const float* gammas = (const float*)d_in[7];
    const float* betas  = (const float*)d_in[8];
    const float* fc1w   = (const float*)d_in[9];
    const float* fc1b   = (const float*)d_in[10];
    const float* fc2w   = (const float*)d_in[11];
    const float* fc2b   = (const float*)d_in[12];
    float* out = (float*)d_out;

    const size_t NF = (size_t)N_NODES * FEATS;     // 12.8M floats (mult of 32)
    // Workspace layout. All bf16 regions start at 32B-aligned offsets by
    // construction (every preceding region size is a multiple of 32 bytes).
    float*  H     = (float*)d_ws;                  // 12.8M f32
    float*  Z     = H + NF;                        // 12.8M f32
    float*  Y     = Z + NF;                        // 12.8M f32
    float*  sum   = Y + NF;                        // 128
    float*  sumsq = sum + FEATS;                   // 128
    float*  scale = sumsq + FEATS;                 // 128
    float*  shift = scale + FEATS;                 // 128
    __bf16* Wh    = (__bf16*)(shift + FEATS);      // 16384 bf16 (32KB)
    __bf16* Wl    = Wh + FEATS * FEATS;            // 16384 bf16
    __bf16* Ah    = Wl + FEATS * FEATS;            // 12.8M bf16 (25.6MB)
    __bf16* Al    = Ah + NF;                       // 12.8M bf16
    float*  psum  = (float*)(Al + NF);             // 100*128
    float*  pcnt  = psum + NUM_GRAPHS * FEATS;     // 100
    float*  hg    = pcnt + NUM_GRAPHS;             // 100*128

    const int n4          = (int)(NF / 4);                 // 3.2M
    const int blkApply    = (n4 + 255) / 256;              // 12500
    const int blkScatter  = (N_EDGES * 32 + 255) / 256;    // 200000
    const int blkGemm     = N_NODES / 16;                  // 6250
    const int blkStats    = 250;
    const int rowsPerBlk  = N_NODES / blkStats;            // 400
    const int blkPool     = (N_NODES * 32 + 255) / 256;    // 12500
    const int blkPrepW    = (FEATS * FEATS + 255) / 256;   // 64

    // Z = H = input features
    init_copy2_kernel<<<blkApply, 256, 0, stream>>>(h_in, H, Z, n4);

    for (int l = 0; l < 4; ++l) {
        const float* Wl_f = (l == 0) ? W0 : (Ws + (size_t)(l - 1) * FEATS * FEATS);
        const float* bl   = bs + (size_t)l * FEATS;
        zero_kernel<<<1, 256, 0, stream>>>(sum, 2 * FEATS);  // sum + sumsq contiguous
        prep_w_kernel<<<blkPrepW, 256, 0, stream>>>(Wl_f, Wh, Wl);
        edge_scatter_kernel<<<blkScatter, 256, 0, stream>>>(H, ei, Z, N_EDGES);
        split_a_kernel<<<blkApply, 256, 0, stream>>>(Z, Ah, Al, n4);
        gemm_bias_kernel<<<blkGemm, 256, 0, stream>>>(Ah, Al, Wh, Wl, bl, Y);
        stats_kernel<<<blkStats, 128, 0, stream>>>(Y, sum, sumsq, N_NODES, rowsPerBlk);
        bn_finalize_kernel<<<1, 128, 0, stream>>>(sum, sumsq,
                                                  gammas + (size_t)l * FEATS,
                                                  betas + (size_t)l * FEATS,
                                                  scale, shift, 1.0f / (float)N_NODES);
        bn_relu_kernel<<<blkApply, 256, 0, stream>>>(Y, scale, shift, H, Z, n4);
    }

    zero_kernel<<<(NUM_GRAPHS * FEATS + NUM_GRAPHS + 255) / 256 + 1, 256, 0, stream>>>(
        psum, NUM_GRAPHS * FEATS + NUM_GRAPHS);  // psum + pcnt contiguous
    pool_scatter_kernel<<<blkPool, 256, 0, stream>>>(H, batch, psum, pcnt, N_NODES);
    pool_div_kernel<<<(NUM_GRAPHS * FEATS + 255) / 256, 256, 0, stream>>>(psum, pcnt, hg);
    head_kernel<<<1, 128, 0, stream>>>(hg, fc1w, fc1b, fc2w, fc2b, out);
}
